// VoxelGNNDiscriminator_61100204753235
// MI455X (gfx1250) — matrix operand order repaired and checked by reading, not compiled, not run on hardware
//
#include <hip/hip_runtime.h>
#include <hip/hip_bf16.h>
#include <stdint.h>

// ---------------------------------------------------------------------------
// VoxelGNNDiscriminator for MI455X (gfx1250): bf16 WMMA GEMMs + atomic GCN agg
// GEMM: BM=256 BN=64 BK=32, double-buffered LDS with ASYNC global->LDS DMA
// for the A tile (no VGPR staging), 8 WMMAs/wave/K-step.
// ---------------------------------------------------------------------------

typedef __attribute__((ext_vector_type(16))) __bf16 v16bf;
typedef __attribute__((ext_vector_type(8)))  __bf16 bf16x8;
typedef __attribute__((ext_vector_type(8)))  float  v8f;

#define NV 131072
#define NE 1048576

static __device__ __forceinline__ unsigned short f32_to_bf16(float f) {
    union { float f; unsigned u; } v; v.f = f;
    unsigned r = v.u + 0x7FFFu + ((v.u >> 16) & 1u);   // round-to-nearest-even
    return (unsigned short)(r >> 16);
}
static __device__ __forceinline__ float bf16_to_f32(unsigned short h) {
    union { unsigned u; float f; } v; v.u = ((unsigned)h) << 16;
    return v.f;
}

// Async DMA of 64 contiguous bytes global -> LDS (4 x B128), no VGPR staging.
// ISA: the instruction offset is added to both the LDS and the global address.
static __device__ __forceinline__ void async_copy64(unsigned lds_addr,
                                                    const unsigned short* gsrc) {
    asm volatile(
        "global_load_async_to_lds_b128 %0, %1, off\n\t"
        "global_load_async_to_lds_b128 %0, %1, off offset:16\n\t"
        "global_load_async_to_lds_b128 %0, %1, off offset:32\n\t"
        "global_load_async_to_lds_b128 %0, %1, off offset:48"
        :: "v"(lds_addr), "v"(gsrc) : "memory");
}
static __device__ __forceinline__ void wait_async() {
    asm volatile("s_wait_asynccnt 0x0" ::: "memory");
}

// ------------------------- type-grouped mean matching ----------------------
__global__ void k_type_stats(const float* __restrict__ lx, const int* __restrict__ lt,
                             float* __restrict__ sums, float* __restrict__ cnt) {
    int i = blockIdx.x;          // 512 local nodes
    int j = threadIdx.x;         // 128 dims
    int t = lt[i];
    atomicAdd(&sums[t * 128 + j], lx[i * 128 + j]);
    if (j == 0) atomicAdd(&cnt[t], 1.0f);
}

__global__ void k_type_mean(const float* __restrict__ sums, const float* __restrict__ cnt,
                            float* __restrict__ mean) {
    int tid = blockIdx.x * blockDim.x + threadIdx.x;   // 1024 = 8*128
    int t = tid >> 7, j = tid & 127;
    float c = cnt[t];
    mean[tid] = (c > 0.0f) ? sums[t * 128 + j] / fmaxf(c, 1.0f) : 0.0f;
    (void)j;
}

// ------------------------- input concat -> bf16 [NV, 288] ------------------
__global__ void k_concat(const float* __restrict__ tmean, const int* __restrict__ vtype,
                         const float* __restrict__ vx, const float* __restrict__ label,
                         unsigned short* __restrict__ X) {
    int idx = blockIdx.x * blockDim.x + threadIdx.x;   // NV*288
    int i = idx / 288, j = idx % 288;
    float v;
    if (j < 128)       v = tmean[vtype[i] * 128 + j];
    else if (j < 256)  v = vx[(size_t)i * 128 + (j - 128)];
    else if (j < 264)  v = label[(size_t)i * 8 + (j - 256)];
    else               v = 0.0f;
    X[idx] = f32_to_bf16(v);
}

// ---- weight f32[din,dout] -> bf16[kpad,npad] (zero padded both dims) ------
__global__ void k_wconv(const float* __restrict__ W, unsigned short* __restrict__ Wb,
                        int din, int dout, int npad) {
    int idx = blockIdx.x * blockDim.x + threadIdx.x;   // kpad*npad
    int k = idx / npad, j = idx % npad;
    Wb[idx] = (k < din && j < dout) ? f32_to_bf16(W[(size_t)k * dout + j])
                                    : (unsigned short)0;
}

// ------------------------- tiled bf16 WMMA GEMM ----------------------------
// C[M,N] = A[M,K](bf16) x B[K,Nb](bf16) (+bias) (+relu).
// M = NV (mult of 256), K % 32 == 0, Nb % 64 == 0 (B zero-padded), N <= Nb.
// 256 threads = 8 waves; wave w owns rows [32w, 32w+32) x 64 cols:
// 2 A-frags x 4 B-frags -> 8 WMMAs per K-step.
// A tile: double-buffered LDS fed by global_load_async_to_lds_b128 (ASYNCcnt).
__global__ __launch_bounds__(256)
void k_gemm_bf16(const unsigned short* __restrict__ A, int lda,
                 const unsigned short* __restrict__ B, int Nb, int N, int K,
                 const float* __restrict__ bias, int relu,
                 float* __restrict__ Cf, unsigned short* __restrict__ Cbf, int ldc) {
    __shared__ unsigned short As[2][256 * 32];   // [m][k] row-major
    __shared__ unsigned short Bs[2][64 * 32];    // [n][k] transposed

    const int tid  = threadIdx.x;
    const int wave = tid >> 5;
    const int lane = tid & 31;
    const int m0   = blockIdx.x * 256;
    const int n0   = blockIdx.y * 64;

    v8f acc[2][4];
    #pragma unroll
    for (int ms = 0; ms < 2; ++ms)
        #pragma unroll
        for (int nt = 0; nt < 4; ++nt)
            acc[ms][nt] = (v8f){0.f,0.f,0.f,0.f,0.f,0.f,0.f,0.f};

    const int arow = tid;                 // A loader: one 32-elem row per thread
    const int bkr  = tid >> 3;            // B loader: 32 k-rows x 8 col-chunks
    const int bc0  = (tid & 7) * 8;
    const int ktiles = K / 32;

    const unsigned short* aptr = A + (size_t)(m0 + arow) * lda;
    const unsigned lds_a0 = (unsigned)(uintptr_t)&As[0][arow * 32];
    const unsigned lds_a1 = (unsigned)(uintptr_t)&As[1][arow * 32];

    // ---- prologue: tile 0 (A via async DMA, B via regs + LDS transpose) ----
    {
        async_copy64(lds_a0, aptr);
        uint4 b0 = *(const uint4*)(B + (size_t)bkr * Nb + n0 + bc0);
        const unsigned short* b8 = (const unsigned short*)&b0;
        #pragma unroll
        for (int i = 0; i < 8; ++i) Bs[0][(bc0 + i) * 32 + bkr] = b8[i];
        wait_async();
    }

    uint4 b_reg = {0, 0, 0, 0};
    for (int t = 0; t < ktiles; ++t) {
        __syncthreads();
        const int cur = t & 1;

        // ---- issue DMA + B load for tile t+1 (overlaps with WMMAs) ----
        if (t + 1 < ktiles) {
            const int k0 = (t + 1) * 32;
            async_copy64(cur ? lds_a0 : lds_a1, aptr + k0);
            b_reg = *(const uint4*)(B + (size_t)(k0 + bkr) * Nb + n0 + bc0);
            if (t + 2 < ktiles)   // global_prefetch_b8 hint for tile t+2
                __builtin_prefetch(aptr + (t + 2) * 32, 0, 1);
        }

        // ---- load all fragments, then back-to-back WMMAs ----
        const int mfr = lane & 15, hk = lane >> 4;
        v16bf afrag[2], bfrag[4];
        #pragma unroll
        for (int ms = 0; ms < 2; ++ms) {
            const unsigned short* ap = &As[cur][(wave * 32 + ms * 16 + mfr) * 32 + hk * 8];
            bf16x8 lo = *(const bf16x8*)(ap);         // K = hk*8 .. +7
            bf16x8 hi = *(const bf16x8*)(ap + 16);    // K = 16+hk*8 .. +7
            afrag[ms] = __builtin_shufflevector(lo, hi,
                           0,1,2,3,4,5,6,7,8,9,10,11,12,13,14,15);
        }
        const int nfr = lane & 15, kb = (lane >> 4) * 8;
        #pragma unroll
        for (int nt = 0; nt < 4; ++nt) {
            const unsigned short* bp = &Bs[cur][(nt * 16 + nfr) * 32 + kb];
            bf16x8 blo = *(const bf16x8*)(bp);
            bf16x8 bhi = *(const bf16x8*)(bp + 16);
            bfrag[nt] = __builtin_shufflevector(blo, bhi,
                           0,1,2,3,4,5,6,7,8,9,10,11,12,13,14,15);
        }

        #pragma unroll
        for (int ms = 0; ms < 2; ++ms)
            #pragma unroll
            for (int nt = 0; nt < 4; ++nt)
                acc[ms][nt] = __builtin_amdgcn_wmma_f32_16x16x32_bf16(
                                  false, afrag[ms], false, bfrag[nt],
                                  (short)0, acc[ms][nt], false, false);

        // ---- publish tile t+1: B regs -> LDS, drain async DMA ----
        if (t + 1 < ktiles) {
            const int nxt = cur ^ 1;
            const unsigned short* b8 = (const unsigned short*)&b_reg;
            #pragma unroll
            for (int i = 0; i < 8; ++i) Bs[nxt][(bc0 + i) * 32 + bkr] = b8[i];
            wait_async();
        }
    }

    // ---- epilogue: D layout lane n = l%16, row = v + 8*(l/16) ----
    const int cbl = n0 + (lane & 15);
    #pragma unroll
    for (int ms = 0; ms < 2; ++ms) {
        const int rbase = m0 + wave * 32 + ms * 16 + (lane >> 4) * 8;
        #pragma unroll
        for (int nt = 0; nt < 4; ++nt) {
            int col = cbl + nt * 16;
            if (col < N) {
                float bv = bias ? bias[col] : 0.0f;
                #pragma unroll
                for (int v = 0; v < 8; ++v) {
                    float x = acc[ms][nt][v] + bv;
                    if (relu) x = fmaxf(x, 0.0f);
                    size_t off = (size_t)(rbase + v) * ldc + col;
                    if (Cf)  Cf[off]  = x;
                    if (Cbf) Cbf[off] = f32_to_bf16(x);
                }
            }
        }
    }
}

// ------------------------- degree / dinv -----------------------------------
__global__ void k_deg(const int* __restrict__ dst, float* __restrict__ deg) {
    int e = blockIdx.x * blockDim.x + threadIdx.x;
    atomicAdd(&deg[dst[e]], 1.0f);
}
__global__ void k_dinv(const float* __restrict__ deg, float* __restrict__ dinv) {
    int i = blockIdx.x * blockDim.x + threadIdx.x;
    dinv[i] = rsqrtf(deg[i] + 1.0f);   // +1 self loop
}

// ------------------------- edge aggregation (bandwidth-bound) --------------
__global__ void k_edge_agg(const int* __restrict__ src, const int* __restrict__ dst,
                           const float* __restrict__ H, const float* __restrict__ dinv,
                           float* __restrict__ AGG, int d, int chunks) {
    long long idx = (long long)blockIdx.x * blockDim.x + threadIdx.x; // NE*chunks
    int e = (int)(idx / chunks);
    int c = (int)(idx % chunks);
    int s = src[e], t = dst[e];
    float w = dinv[s] * dinv[t];
    float4 h = *(const float4*)&H[(size_t)s * d + c * 4];
    float* out = &AGG[(size_t)t * d + c * 4];
    atomicAdd(out + 0, h.x * w);
    atomicAdd(out + 1, h.y * w);
    atomicAdd(out + 2, h.z * w);
    atomicAdd(out + 3, h.w * w);
}

// ---- fused: self-loop + bias into AGG, column sum / sum-of-squares --------
__global__ void k_self_stats(float* __restrict__ AGG, const float* __restrict__ H,
                             const float* __restrict__ dinv, const float* __restrict__ bias,
                             float* __restrict__ cs, float* __restrict__ cs2, int d) {
    int tid = blockIdx.x * blockDim.x + threadIdx.x;   // d * 1024 threads
    int j = tid % d, r0 = tid / d;
    float b = bias[j], s1 = 0.f, s2 = 0.f;
    for (int r = r0; r < NV; r += 1024) {
        float di = dinv[r];
        size_t o = (size_t)r * d + j;
        float v = AGG[o] + H[o] * di * di + b;
        AGG[o] = v;
        s1 += v; s2 += v * v;
    }
    atomicAdd(&cs[j], s1);
    atomicAdd(&cs2[j], s2);
}

// ---- GraphNorm coefficients: var(x-a*m) = E[x^2] + (a^2-2a)m^2 ------------
__global__ void k_norm_coef(const float* __restrict__ cs, const float* __restrict__ cs2,
                            const float* __restrict__ gw, const float* __restrict__ gb,
                            const float* __restrict__ ga,
                            float* __restrict__ scale, float* __restrict__ shift, int d) {
    int j = threadIdx.x;
    if (j < d) {
        float m = cs[j] / (float)NV;
        float a = ga[j];
        float var = cs2[j] / (float)NV + (a * a - 2.0f * a) * m * m;
        float s = gw[j] * rsqrtf(var + 1e-5f);
        scale[j] = s;
        shift[j] = gb[j] - s * a * m;
    }
}

__global__ void k_norm_apply(const float* __restrict__ AGG, const float* __restrict__ scale,
                             const float* __restrict__ shift, unsigned short* __restrict__ X,
                             int d) {
    long long idx = (long long)blockIdx.x * blockDim.x + threadIdx.x;  // NV*d
    int j = (int)(idx % d);
    float y = fmaxf(scale[j] * AGG[idx] + shift[j], 0.0f);
    X[idx] = f32_to_bf16(y);
}

// ------------------------- final 32->1 projection --------------------------
__global__ void k_final(const unsigned short* __restrict__ X, const float* __restrict__ W,
                        const float* __restrict__ b, float* __restrict__ out) {
    int i = blockIdx.x * blockDim.x + threadIdx.x;
    float s = b[0];
    #pragma unroll
    for (int j = 0; j < 32; ++j) s += bf16_to_f32(X[(size_t)i * 32 + j]) * W[j];
    out[i] = s;
}

// ===========================================================================
extern "C" void kernel_launch(void* const* d_in, const int* in_sizes, int n_in,
                              void* d_out, int out_size, void* d_ws, size_t ws_size,
                              hipStream_t stream) {
    (void)in_sizes; (void)n_in; (void)out_size; (void)ws_size;

    const float* local_x    = (const float*)d_in[0];
    const int*   local_type = (const int*)d_in[1];
    const float* voxel_x    = (const float*)d_in[2];
    const int*   voxel_type = (const int*)d_in[3];
    const int*   edge_index = (const int*)d_in[4];
    const float* label_hard = (const float*)d_in[5];
    const float* enc_w1 = (const float*)d_in[6];
    const float* enc_b1 = (const float*)d_in[7];
    const float* enc_w2 = (const float*)d_in[8];
    const float* enc_b2 = (const float*)d_in[9];
    const float* convW[4] = {(const float*)d_in[10], (const float*)d_in[12],
                             (const float*)d_in[14], (const float*)d_in[16]};
    const float* convB[4] = {(const float*)d_in[11], (const float*)d_in[13],
                             (const float*)d_in[15], (const float*)d_in[17]};
    const float* gnW[4] = {(const float*)d_in[18], (const float*)d_in[21],
                           (const float*)d_in[24], (const float*)d_in[27]};
    const float* gnB[4] = {(const float*)d_in[19], (const float*)d_in[22],
                           (const float*)d_in[25], (const float*)d_in[28]};
    const float* gnA[4] = {(const float*)d_in[20], (const float*)d_in[23],
                           (const float*)d_in[26], (const float*)d_in[29]};
    const float* decW[4] = {(const float*)d_in[30], (const float*)d_in[32],
                            (const float*)d_in[34], (const float*)d_in[36]};
    const float* decB[4] = {(const float*)d_in[31], (const float*)d_in[33],
                            (const float*)d_in[35], (const float*)d_in[37]};
    float* out = (float*)d_out;

    // ---- workspace layout ----
    char* ws = (char*)d_ws;
    size_t off = 0;
    auto alloc = [&](size_t bytes) -> void* {
        void* p = ws + off;
        off = (off + bytes + 255) & ~(size_t)255;
        return p;
    };
    float* t_sums  = (float*)alloc(8 * 128 * 4);
    float* t_cnt   = (float*)alloc(8 * 4);
    float* t_mean  = (float*)alloc(8 * 128 * 4);
    float* deg     = (float*)alloc((size_t)NV * 4);
    float* dinv    = (float*)alloc((size_t)NV * 4);
    float* colsum  = (float*)alloc(256 * 4);
    float* colsum2 = (float*)alloc(256 * 4);
    float* nscale  = (float*)alloc(256 * 4);
    float* nshift  = (float*)alloc(256 * 4);

    // bf16 weights, K padded to 32-mult, N padded to 64-mult (zero fill)
    unsigned short* encW1p = (unsigned short*)alloc((size_t)288 * 256 * 2);
    unsigned short* encW2p = (unsigned short*)alloc((size_t)256 * 256 * 2);
    int cdin[4]  = {256, 128, 64, 128};
    int cdout[4] = {128, 64, 128, 256};
    unsigned short* convWp[4];
    for (int l = 0; l < 4; ++l)
        convWp[l] = (unsigned short*)alloc((size_t)cdin[l] * cdout[l] * 2);
    int ddin[3]  = {256, 128, 64};
    int ddout[3] = {128, 64, 32};
    int ddpad[3] = {128, 64, 64};
    unsigned short* decWp[3];
    for (int l = 0; l < 3; ++l)
        decWp[l] = (unsigned short*)alloc((size_t)ddin[l] * ddpad[l] * 2);

    unsigned short* XA = (unsigned short*)alloc((size_t)NV * 288 * 2);
    unsigned short* XB = (unsigned short*)alloc((size_t)NV * 288 * 2);
    float* H   = (float*)alloc((size_t)NV * 256 * 4);
    float* AGG = (float*)alloc((size_t)NV * 256 * 4);

    auto gemm = [&](const unsigned short* A, int lda, const unsigned short* B,
                    int Nb, int N, int K, const float* bias, int relu,
                    float* Cf, unsigned short* Cbf, int ldc) {
        dim3 grid(NV / 256, Nb / 64);
        hipLaunchKernelGGL(k_gemm_bf16, grid, dim3(256), 0, stream,
                           A, lda, B, Nb, N, K, bias, relu, Cf, Cbf, ldc);
    };

    // ---- 1. type-grouped mean matching ----
    hipMemsetAsync(t_sums, 0, 8 * 128 * 4, stream);
    hipMemsetAsync(t_cnt, 0, 8 * 4, stream);
    hipLaunchKernelGGL(k_type_stats, dim3(512), dim3(128), 0, stream,
                       local_x, local_type, t_sums, t_cnt);
    hipLaunchKernelGGL(k_type_mean, dim3(4), dim3(256), 0, stream, t_sums, t_cnt, t_mean);

    // ---- 2. convert weights to padded bf16 ----
    auto wconv = [&](const float* W, unsigned short* Wb, int din, int dout,
                     int kpad, int npad) {
        hipLaunchKernelGGL(k_wconv, dim3((kpad * npad) / 256), dim3(256), 0, stream,
                           W, Wb, din, dout, npad);
    };
    wconv(enc_w1, encW1p, 264, 256, 288, 256);
    wconv(enc_w2, encW2p, 256, 256, 256, 256);
    for (int l = 0; l < 4; ++l) wconv(convW[l], convWp[l], cdin[l], cdout[l], cdin[l], cdout[l]);
    for (int l = 0; l < 3; ++l) wconv(decW[l], decWp[l], ddin[l], ddout[l], ddin[l], ddpad[l]);

    // ---- 3. degree / dinv (shared by all conv layers) ----
    hipMemsetAsync(deg, 0, (size_t)NV * 4, stream);
    hipLaunchKernelGGL(k_deg, dim3(NE / 256), dim3(256), 0, stream, edge_index + NE, deg);
    hipLaunchKernelGGL(k_dinv, dim3(NV / 256), dim3(256), 0, stream, deg, dinv);

    // ---- 4. concat input [NV, 288] bf16 ----
    hipLaunchKernelGGL(k_concat, dim3((size_t)NV * 288 / 256), dim3(256), 0, stream,
                       t_mean, voxel_type, voxel_x, label_hard, XA);

    unsigned short* Xcur = XA;
    unsigned short* Xnxt = XB;

    // ---- 5. MLP encoder ----
    gemm(Xcur, 288, encW1p, 256, 256, 288, enc_b1, 1, nullptr, Xnxt, 256);
    { unsigned short* t = Xcur; Xcur = Xnxt; Xnxt = t; }
    gemm(Xcur, 256, encW2p, 256, 256, 256, enc_b2, 1, nullptr, Xnxt, 256);
    { unsigned short* t = Xcur; Xcur = Xnxt; Xnxt = t; }

    // ---- 6. GCN conv + GraphNorm + ReLU x4 ----
    for (int l = 0; l < 4; ++l) {
        int din = cdin[l], d = cdout[l];
        gemm(Xcur, din, convWp[l], d, d, din, nullptr, 0, H, nullptr, d);
        hipMemsetAsync(AGG, 0, (size_t)NV * d * 4, stream);
        hipMemsetAsync(colsum, 0, d * 4, stream);
        hipMemsetAsync(colsum2, 0, d * 4, stream);
        int chunks = d / 4;
        hipLaunchKernelGGL(k_edge_agg, dim3((size_t)NE * chunks / 256), dim3(256), 0, stream,
                           edge_index, edge_index + NE, H, dinv, AGG, d, chunks);
        hipLaunchKernelGGL(k_self_stats, dim3(d * 4), dim3(256), 0, stream,
                           AGG, H, dinv, convB[l], colsum, colsum2, d);
        hipLaunchKernelGGL(k_norm_coef, dim3(1), dim3(256), 0, stream,
                           colsum, colsum2, gnW[l], gnB[l], gnA[l], nscale, nshift, d);
        hipLaunchKernelGGL(k_norm_apply, dim3((size_t)NV * d / 256), dim3(256), 0, stream,
                           AGG, nscale, nshift, Xnxt, d);
        { unsigned short* t = Xcur; Xcur = Xnxt; Xnxt = t; }
    }

    // ---- 7. decoder MLP ----
    for (int l = 0; l < 3; ++l) {
        gemm(Xcur, ddin[l], decWp[l], ddpad[l], ddout[l], ddin[l], decB[l], 1,
             nullptr, Xnxt, ddout[l]);
        { unsigned short* t = Xcur; Xcur = Xnxt; Xnxt = t; }
    }
    hipLaunchKernelGGL(k_final, dim3(NV / 256), dim3(256), 0, stream,
                       Xcur, decW[3], decB[3], out);
}
